// NTXEntLossAEFoster_32014686224384
// MI455X (gfx1250) — matrix operand order repaired and checked by reading, not compiled, not run on hardware
//
#include <hip/hip_runtime.h>

// ---------------------------------------------------------------------------
// NT-Xent loss, MI455X (gfx1250, wave32, WMMA).
//   loss = (1/2N) * sum_i [ log(Sa[i]-e^2) + log(Sb[i]-e^2) - 2*d[i] ]
//   Sa = rowsum exp(2*a@bT) + rowsum exp(2*a@aT)   (diag of a@aT removed as e^2)
//   Sb = rowsum exp(2*b@aT) + rowsum exp(2*b@bT)
//   d[i] = 2*cos(a_i, b_i), computed in fp32 from raw inputs.
// GEMMs in f16 (normalized operands) with f32 WMMA accumulation, exp-rowsum
// fused into the epilogue; sim matrices are never materialized.
// GEMM inner loop is software double-buffered: fragments for K-step k+32 are
// loaded while the 8 WMMAs of K-step k execute (2 A-frags x 4 B-frags per
// wave = 32x64 wave tile -> 12 b128 loads per 8 WMMAs).
// ---------------------------------------------------------------------------

#define DDIM 512
#define INV_T 2.0f
#define EPSN 1e-8f

typedef __attribute__((ext_vector_type(16))) _Float16 v16h;
typedef __attribute__((ext_vector_type(8)))  _Float16 v8h;
typedef __attribute__((ext_vector_type(8)))  float    v8f;

union V16 { v16h v; v8h h[2]; };

__device__ __forceinline__ float wave32_allsum(float s) {
  s += __shfl_xor(s, 1, 32);
  s += __shfl_xor(s, 2, 32);
  s += __shfl_xor(s, 4, 32);
  s += __shfl_xor(s, 8, 32);
  s += __shfl_xor(s, 16, 32);
  return s;
}

// ---- zero the row-sum accumulators ----------------------------------------
__global__ void zero_kernel(float* __restrict__ p, int n) {
  int i = blockIdx.x * blockDim.x + threadIdx.x;
  if (i < n) p[i] = 0.0f;
}

// ---- L2-normalize rows and convert to f16 (one wave per row, 8 rows/block) -
__global__ __launch_bounds__(256)
void nrm_kernel(const float* __restrict__ srcA, const float* __restrict__ srcB,
                _Float16* __restrict__ dstA, _Float16* __restrict__ dstB, int N) {
  const float* src = (blockIdx.y == 0) ? srcA : srcB;
  _Float16*    dst = (blockIdx.y == 0) ? dstA : dstB;
  const int wave = threadIdx.x >> 5, lane = threadIdx.x & 31;
  const int row  = blockIdx.x * 8 + wave;
  if (row >= N) return;
  const float4* p = (const float4*)(src + (size_t)row * DDIM + lane * 16);
  float4 x[4];
  float s = 0.0f;
#pragma unroll
  for (int i = 0; i < 4; ++i) {
    x[i] = p[i];
    s += x[i].x * x[i].x + x[i].y * x[i].y + x[i].z * x[i].z + x[i].w * x[i].w;
  }
  s = wave32_allsum(s);
  const float inv = 1.0f / fmaxf(sqrtf(s), EPSN);
  v8h o0, o1;
  o0[0] = (_Float16)(x[0].x * inv); o0[1] = (_Float16)(x[0].y * inv);
  o0[2] = (_Float16)(x[0].z * inv); o0[3] = (_Float16)(x[0].w * inv);
  o0[4] = (_Float16)(x[1].x * inv); o0[5] = (_Float16)(x[1].y * inv);
  o0[6] = (_Float16)(x[1].z * inv); o0[7] = (_Float16)(x[1].w * inv);
  o1[0] = (_Float16)(x[2].x * inv); o1[1] = (_Float16)(x[2].y * inv);
  o1[2] = (_Float16)(x[2].z * inv); o1[3] = (_Float16)(x[2].w * inv);
  o1[4] = (_Float16)(x[3].x * inv); o1[5] = (_Float16)(x[3].y * inv);
  o1[6] = (_Float16)(x[3].z * inv); o1[7] = (_Float16)(x[3].w * inv);
  v8h* q = (v8h*)(dst + (size_t)row * DDIM + lane * 16);
  q[0] = o0;
  q[1] = o1;
}

// ---- exact fp32 target logits d[i] = 2*cos(a_i, b_i) -----------------------
__global__ __launch_bounds__(256)
void diag_kernel(const float* __restrict__ A, const float* __restrict__ B,
                 float* __restrict__ dg, int N) {
  const int wave = threadIdx.x >> 5, lane = threadIdx.x & 31;
  const int row  = blockIdx.x * 8 + wave;
  if (row >= N) return;
  const float4* pa = (const float4*)(A + (size_t)row * DDIM + lane * 16);
  const float4* pb = (const float4*)(B + (size_t)row * DDIM + lane * 16);
  float ab = 0.0f, aa = 0.0f, bb = 0.0f;
#pragma unroll
  for (int i = 0; i < 4; ++i) {
    float4 a = pa[i], b = pb[i];
    ab += a.x * b.x + a.y * b.y + a.z * b.z + a.w * b.w;
    aa += a.x * a.x + a.y * a.y + a.z * a.z + a.w * a.w;
    bb += b.x * b.x + b.y * b.y + b.z * b.z + b.w * b.w;
  }
  ab = wave32_allsum(ab);
  aa = wave32_allsum(aa);
  bb = wave32_allsum(bb);
  if (lane == 0)
    dg[row] = INV_T * ab / (fmaxf(sqrtf(aa), EPSN) * fmaxf(sqrtf(bb), EPSN));
}

// ---- fused GEMM + exp-rowsum ----------------------------------------------
// blockIdx.z: 0:(a,b)->accA 1:(a,a)->accA 2:(b,a)->accB 3:(b,b)->accB
// block = 8 waves, 128x128 tile; wave = 32 rows x 64 cols (2x4 WMMA tiles).

__device__ __forceinline__ void load_frags(V16 af[2], V16 bf[4],
                                           const _Float16* __restrict__ arow0,
                                           const _Float16* __restrict__ arow1,
                                           const _Float16* __restrict__ ybase,
                                           int k, int half) {
  // A 16x32 f16 fragment: lanes 0-15 hold K {k..k+15},
  // lanes 16-31 hold K {k+16..k+31} for the same 16 rows.
  af[0].h[0] = *(const v8h*)(arow0 + k + half * 16);
  af[0].h[1] = *(const v8h*)(arow0 + k + 8 + half * 16);
  af[1].h[0] = *(const v8h*)(arow1 + k + half * 16);
  af[1].h[1] = *(const v8h*)(arow1 + k + 8 + half * 16);
  // B 32x16 f16 fragment: lane holds column (nBase+t*16+lr), 16 contiguous
  // K values starting at k + half*16 (row-major Y, no transpose needed).
#pragma unroll
  for (int t = 0; t < 4; ++t) {
    bf[t].h[0] = *(const v8h*)(ybase + t * (16 * DDIM) + k);
    bf[t].h[1] = *(const v8h*)(ybase + t * (16 * DDIM) + k + 8);
  }
}

__device__ __forceinline__ void mm_step(v8f acc[2][4], V16 af[2], V16 bf[4]) {
#pragma unroll
  for (int i = 0; i < 2; ++i)
#pragma unroll
    for (int t = 0; t < 4; ++t)
      acc[i][t] = __builtin_amdgcn_wmma_f32_16x16x32_f16(
          false, af[i].v, false, bf[t].v, (short)0, acc[i][t], false, false);
}

__global__ __launch_bounds__(256)
void gemm_expsum_kernel(const _Float16* __restrict__ Ah,
                        const _Float16* __restrict__ Bh,
                        float* __restrict__ accA, float* __restrict__ accB, int N) {
  const int p = blockIdx.z;
  const _Float16* X = (p < 2) ? Ah : Bh;
  const _Float16* Y = (p == 1 || p == 2) ? Ah : Bh;
  float* acc_out    = (p < 2) ? accA : accB;

  const int wave = threadIdx.x >> 5;
  const int lane = threadIdx.x & 31;
  const int half = lane >> 4;   // K-half selector per WMMA operand layout
  const int lr   = lane & 15;   // row/col within 16-tile

  const int rw = wave >> 1;     // 0..3 : row group
  const int cw = wave & 1;      // 0..1 : col group
  const int mBase = blockIdx.x * 128 + rw * 32;
  const int nBase = blockIdx.y * 128 + cw * 64;

  const _Float16* arow0 = X + (size_t)(mBase + lr) * DDIM;
  const _Float16* arow1 = arow0 + 16 * DDIM;
  const _Float16* ybase = Y + (size_t)(nBase + lr) * DDIM + half * 16;

  v8f acc[2][4];
#pragma unroll
  for (int i = 0; i < 2; ++i)
#pragma unroll
    for (int t = 0; t < 4; ++t) acc[i][t] = (v8f){0, 0, 0, 0, 0, 0, 0, 0};

  // Software double-buffered pipeline over K (two 32-wide steps per round).
  V16 a0[2], b0[4], a1[2], b1[4];
  load_frags(a0, b0, arow0, arow1, ybase, 0, half);
#pragma unroll
  for (int k = 0; k < DDIM; k += 64) {
    load_frags(a1, b1, arow0, arow1, ybase, k + 32, half);  // prefetch next
    mm_step(acc, a0, b0);                                   // compute current
    if (k + 64 < DDIM)
      load_frags(a0, b0, arow0, arow1, ybase, k + 64, half);
    mm_step(acc, a1, b1);
  }

  // Epilogue: C/D layout: VGPR r, lanes 0-15 -> (M=r, N=lane),
  // lanes 16-31 -> (M=r+8, N=lane-16). Row sum of exp(2*c) per output row.
#pragma unroll
  for (int i = 0; i < 2; ++i) {
#pragma unroll
    for (int r = 0; r < 8; ++r) {
      float e = 0.0f;
#pragma unroll
      for (int t = 0; t < 4; ++t) e += __expf(INV_T * acc[i][t][r]);
      e += __shfl_xor(e, 1, 32);
      e += __shfl_xor(e, 2, 32);
      e += __shfl_xor(e, 4, 32);
      e += __shfl_xor(e, 8, 32);
      if (lr == 0) atomicAdd(&acc_out[mBase + i * 16 + half * 8 + r], e);
    }
  }
}

// ---- final reduction to the scalar loss -----------------------------------
__global__ __launch_bounds__(256)
void finalize_kernel(const float* __restrict__ accA, const float* __restrict__ accB,
                     const float* __restrict__ dg, float* __restrict__ out, int N) {
  __shared__ float red[256];
  const float E2 = 7.38905609893065f;  // exp(inv_t * 1.0) = masked diagonal term
  float s = 0.0f;
  for (int i = threadIdx.x; i < N; i += 256)
    s += logf(accA[i] - E2) + logf(accB[i] - E2) - 2.0f * dg[i];
  red[threadIdx.x] = s;
  __syncthreads();
  for (int off = 128; off > 0; off >>= 1) {
    if (threadIdx.x < off) red[threadIdx.x] += red[threadIdx.x + off];
    __syncthreads();
  }
  if (threadIdx.x == 0) out[0] = red[0] / (2.0f * (float)N);
}

extern "C" void kernel_launch(void* const* d_in, const int* in_sizes, int n_in,
                              void* d_out, int out_size, void* d_ws, size_t ws_size,
                              hipStream_t stream) {
  const float* a = (const float*)d_in[0];
  const float* b = (const float*)d_in[1];
  float* out = (float*)d_out;
  const int N = in_sizes[0] / DDIM;  // 4096

  char* ws = (char*)d_ws;
  const size_t fbytes = (size_t)N * DDIM * sizeof(_Float16);  // 4 MB each
  _Float16* Ah = (_Float16*)ws;
  _Float16* Bh = (_Float16*)(ws + fbytes);
  float* accA = (float*)(ws + 2 * fbytes);
  float* accB = accA + N;
  float* dg   = accB + N;

  zero_kernel<<<(2 * N + 255) / 256, 256, 0, stream>>>(accA, 2 * N);
  nrm_kernel<<<dim3((N + 7) / 8, 2), 256, 0, stream>>>(a, b, Ah, Bh, N);
  diag_kernel<<<(N + 7) / 8, 256, 0, stream>>>(a, b, dg, N);
  gemm_expsum_kernel<<<dim3(N / 128, N / 128, 4), 256, 0, stream>>>(Ah, Bh, accA, accB, N);
  finalize_kernel<<<1, 256, 0, stream>>>(accA, accB, dg, out, N);
}
